// EncoderLayer_63118839382221
// MI455X (gfx1250) — matrix operand order repaired and checked by reading, not compiled, not run on hardware
//
#include <hip/hip_runtime.h>

// ---------------------------------------------------------------------------
// Problem constants (from reference): F=256, N=400, C=128, H=8, E=16, M=128
// ---------------------------------------------------------------------------
#define FF   256
#define NN   400
#define CC   128
#define HH   8
#define EE   16
#define MM   128
#define ROWS (FF * NN)   // 102400 flattened (f,n) rows

typedef __attribute__((ext_vector_type(16))) __bf16          v16bf;
typedef __attribute__((ext_vector_type(8)))  float           v8f;
typedef __attribute__((ext_vector_type(16))) unsigned short  v16u;
typedef __attribute__((ext_vector_type(4)))  unsigned int    u32x4;
typedef __attribute__((ext_vector_type(8)))  int             i32x8;
typedef __attribute__((ext_vector_type(4)))  int             i32x4;

union FragU { v16u u; v16bf b; };

// Hardware f32 -> bf16 (v_cvt_*_bf16_f32, RNE) instead of manual bit twiddling
__device__ __forceinline__ unsigned short f2bf(float x) {
    union { __bf16 h; unsigned short u; } cv;
    cv.h = (__bf16)x;
    return cv.u;
}

__device__ __forceinline__ v8f vzero8() {
    v8f z;
#pragma unroll
    for (int i = 0; i < 8; ++i) z[i] = 0.0f;
    return z;
}

// D = A(16x32 bf16) * B(32x16 bf16) + C(16x16 f32)
__device__ __forceinline__ v8f wmma_bf16(v16u a, v16u b, v8f c) {
    FragU fa, fb; fa.u = a; fb.u = b;
    return __builtin_amdgcn_wmma_f32_16x16x32_bf16(
        false, fa.b, false, fb.b, (short)0, c, false, false);
}

// ---------------------------------------------------------------------------
// Tensor Data Mover: 1-D copy of nelem bf16 elements global -> LDS.
// D# per ISA 8.3/8.4: group0 {count=1, lds_addr, global_addr, type=2},
// group1 {data_size=2B, tensor_dim0=nelem, tensor_dim1=1, tile_dim0=nelem, 1-D}.
// ---------------------------------------------------------------------------
__device__ __forceinline__ void tdm_load_1d(unsigned int lds_off,
                                            const void* gptr,
                                            unsigned int nelem) {
    unsigned long long ga = (unsigned long long)gptr;
    u32x4 g0;
    g0[0] = 1u;                                            // count=1 (valid user D#)
    g0[1] = lds_off;                                       // lds_addr (bytes)
    g0[2] = (unsigned int)(ga & 0xFFFFFFFFu);              // global_addr[31:0]
    g0[3] = (unsigned int)((ga >> 32) & 0x01FFFFFFu)       // global_addr[56:32]
          | 0x80000000u;                                   // type = 2 ("image")
    i32x8 g1;
    g1[0] = (int)(1u << 16);                               // data_size = 2 bytes
    g1[1] = (int)((nelem & 0xFFFFu) << 16);                // tensor_dim0[15:0]
    g1[2] = (int)(((nelem >> 16) & 0xFFFFu) | (1u << 16)); // tensor_dim0 hi | tensor_dim1=1
    g1[3] = (int)((nelem & 0xFFFFu) << 16);                // tile_dim0
    g1[4] = 0;                                             // tile_dim1/2 = 0 (1-D)
    g1[5] = (int)nelem;                                    // tensor_dim0_stride lo
    g1[6] = 0;
    g1[7] = 0;
    i32x4 z4; z4[0] = z4[1] = z4[2] = z4[3] = 0;
#if defined(__clang_major__) && (__clang_major__ >= 23)
    i32x8 z8;
#pragma unroll
    for (int i = 0; i < 8; ++i) z8[i] = 0;
    __builtin_amdgcn_tensor_load_to_lds(g0, g1, z4, z4, z8, 0);
#else
    __builtin_amdgcn_tensor_load_to_lds(g0, g1, z4, z4, 0);
#endif
}

// ---------------------------------------------------------------------------
// Fragment loaders.
// A layout (16-bit 16x32): lane L holds row (L&15); half t -> K = (L>>4)*16 + t (contiguous, same as B).
// B layout (16-bit 32x16): lane L holds column (L&15); half t -> K = (L>>4)*16 + t.
// ---------------------------------------------------------------------------
__device__ __forceinline__ v16u fragA_bf(const unsigned short* p, int ld, int row0, int k0, int lane) {
    const unsigned short* q = p + (size_t)(row0 + (lane & 15)) * ld + k0 + ((lane >> 4) << 4);
    v16u r;
#pragma unroll
    for (int t = 0; t < 8; ++t) r[t] = q[t];
#pragma unroll
    for (int t = 0; t < 8; ++t) r[t + 8] = q[t + 8];
    return r;
}

// B from bf16 weight matrix stored (out, in): contiguous 16 along k
__device__ __forceinline__ v16u fragB_bf(const unsigned short* p, int ld, int col0, int k0, int lane) {
    const unsigned short* q = p + (size_t)(col0 + (lane & 15)) * ld + k0 + ((lane >> 4) << 4);
    v16u r;
#pragma unroll
    for (int t = 0; t < 16; ++t) r[t] = q[t];
    return r;
}

// A operand with K = E = 16 (upper 16 K zero-padded), rows in LDS, stride 16 halves
__device__ __forceinline__ v16u fragA_e16(const unsigned short* lds, int row0, int lane) {
    const unsigned short* q = lds + (row0 + (lane & 15)) * EE;
    v16u r;
#pragma unroll
    for (int t = 0; t < 16; ++t) r[t] = (lane < 16) ? q[t] : (unsigned short)0;
    return r;
}

// B operand whose columns are 16-wide rows in LDS (k/emb as B^T), K = 16 valid
__device__ __forceinline__ v16u fragB_e16(const unsigned short* lds, int row0, int lane) {
    v16u r;
    if (lane < 16) {
        const unsigned short* q = lds + (row0 + lane) * EE;
#pragma unroll
        for (int t = 0; t < 16; ++t) r[t] = q[t];
    } else {
#pragma unroll
        for (int t = 0; t < 16; ++t) r[t] = 0;   // K = 16..31 -> zero
    }
    return r;
}

// B operand = V chunk (32 s-rows x 16 e-cols), V row-major [f][e] in LDS
__device__ __forceinline__ v16u fragB_v(const unsigned short* vlds, int s0, int lane) {
    int e  = lane & 15;
    int kb = (lane >> 4) << 4;
    v16u r;
#pragma unroll
    for (int t = 0; t < 16; ++t) r[t] = vlds[(s0 + kb + t) * EE + e];
    return r;
}

// A operand = probability chunk from bf16 LDS staging (16 x 32, stride 32 halves)
__device__ __forceinline__ v16u fragA_p(const unsigned short* st, int lane) {
    const unsigned short* q = st + (lane & 15) * 32 + ((lane >> 4) << 4);
    v16u r;
#pragma unroll
    for (int t = 0; t < 8; ++t) r[t] = q[t];
#pragma unroll
    for (int t = 0; t < 8; ++t) r[t + 8] = q[t + 8];
    return r;
}

// ---------------------------------------------------------------------------
// Setup: one-pass f32 -> bf16 pre-conversion (hoists cvt out of GEMM loops)
// ---------------------------------------------------------------------------
__global__ __launch_bounds__(256) void k_f2bf(
    const float* __restrict__ s, unsigned short* __restrict__ d, int nelem) {
    int i = blockIdx.x * 256 + threadIdx.x;
    if (i < nelem) d[i] = f2bf(s[i]);
}

// ---------------------------------------------------------------------------
// Kernel 0: tree positional projections  tk/tv[n,o] = sum_c tree[n,c] * W[c,o]
// ---------------------------------------------------------------------------
__global__ __launch_bounds__(128) void k_tree(
    const float* __restrict__ tree, const float* __restrict__ Wk,
    const float* __restrict__ Wv, float* __restrict__ tk, float* __restrict__ tv) {
    int n = blockIdx.x, o = threadIdx.x;
    float a = 0.f, b = 0.f;
#pragma unroll 8
    for (int c = 0; c < CC; ++c) {
        float t = tree[n * CC + c];
        a += t * Wk[c * CC + o];
        b += t * Wv[c * CC + o];
    }
    tk[n * CC + o] = a;
    tv[n * CC + o] = b;
}

// ---------------------------------------------------------------------------
// Kernel 1: QKV projection (102400 x 384 x 128), fused tree add,
//           writes bf16 q/k/v in [n][h][f][e] head layout.
//           grid 6400, block 256: each wave keeps A in VGPRs, does 3 col tiles.
// ---------------------------------------------------------------------------
__global__ __launch_bounds__(256) void k_qkv(
    const unsigned short* __restrict__ xb, const unsigned short* __restrict__ Wq,
    const float* __restrict__ tk, const float* __restrict__ tv,
    unsigned short* __restrict__ qg, unsigned short* __restrict__ kg,
    unsigned short* __restrict__ vg) {
    int tid = threadIdx.x, lane = tid & 31, w = tid >> 5;
    int rb = blockIdx.x << 4;
    int mrow = (lane >> 4) << 3, col = lane & 15;

    v16u a[4];
#pragma unroll
    for (int kk = 0; kk < 4; ++kk) a[kk] = fragA_bf(xb, CC, rb, kk * 32, lane);

#pragma unroll
    for (int j = 0; j < 3; ++j) {
        int ob = (w + j * 8) << 4;                  // 24 col tiles over 8 waves
        v8f acc = vzero8();
#pragma unroll
        for (int kk = 0; kk < 4; ++kk)
            acc = wmma_bf16(a[kk], fragB_bf(Wq, CC, ob, kk * 32, lane), acc);

        int o = ob + col;
        int which = o >> 7, cc = o & 127, hh = cc >> 4, e = cc & 15;
        unsigned short* dst = (which == 0) ? qg : ((which == 1) ? kg : vg);
        const float* add = (which == 1) ? tk : ((which == 2) ? tv : nullptr);
#pragma unroll
        for (int r = 0; r < 8; ++r) {
            int row = rb + mrow + r;
            int f = row / NN, n = row - f * NN;
            float v = acc[r];
            if (add) v += add[n * CC + cc];
            dst[(((size_t)n * HH + hh) * FF + f) * EE + e] = f2bf(v);
        }
    }
}

// ---------------------------------------------------------------------------
// Kernel 2: fused attention, one (n,h) per block. TDM fills LDS; 4 waves,
// each owns 4 l-tiles.
//   scores: S = QK^T + gather(Q emb^T) + gather(K emb^T)   (5 WMMA / 16x16 tile)
//   softmax in registers (shfl_xor within 16-lane halves, wave32)
//   O = P V   (8 WMMA per l-tile, K=32 chunks, bf16 staging in LDS)
// ---------------------------------------------------------------------------
__global__ __launch_bounds__(128) void k_attn(
    const unsigned short* __restrict__ qg, const unsigned short* __restrict__ kg,
    const unsigned short* __restrict__ vg, const unsigned short* __restrict__ embb,
    float* __restrict__ y) {
    __shared__ __align__(16) unsigned short s_q[FF * EE];
    __shared__ __align__(16) unsigned short s_k[FF * EE];
    __shared__ __align__(16) unsigned short s_v[FF * EE];
    __shared__ __align__(16) unsigned short s_emb[(2 * MM + 1) * EE];
    __shared__ __align__(16) float s_stg [4][16 * 32];   // per-wave QE staging
    __shared__ __align__(16) float s_stg2[4][16 * 32];   // per-wave KE staging

    int nh = blockIdx.x;                    // 0..3199
    int n = nh >> 3, h = nh & 7;
    int tid = threadIdx.x, lane = tid & 31, w = tid >> 5;
    size_t base = (size_t)nh * FF * EE;

    // --- Tensor Data Mover: async DMA of q/k/v (8KB each) + emb into LDS ---
    if (tid == 0) {
        tdm_load_1d((unsigned int)(size_t)(void*)s_q,   qg + base, FF * EE);
        tdm_load_1d((unsigned int)(size_t)(void*)s_k,   kg + base, FF * EE);
        tdm_load_1d((unsigned int)(size_t)(void*)s_v,   vg + base, FF * EE);
        tdm_load_1d((unsigned int)(size_t)(void*)s_emb, embb, (2 * MM + 1) * EE);
        __builtin_amdgcn_s_wait_tensorcnt((short)0);
    }
    __syncthreads();

    float* stg  = &s_stg[w][0];
    float* stg2 = &s_stg2[w][0];
    unsigned short* pstg = (unsigned short*)stg;         // bf16 view for P staging
    const int mrow = (lane >> 4) << 3;
    const int col  = lane & 15;
    const float scale = 0.25f;              // 1/sqrt(E)

    for (int lt = w; lt < 16; lt += 4) {
        int lb = lt << 4;
        v16u aQ = fragA_e16(s_q, lb, lane);
        v8f S[16];

#pragma unroll
        for (int st = 0; st < 16; ++st) {
            int sb = st << 4;
            v16u bK = fragB_e16(s_k, sb, lane);
            v8f acc = wmma_bf16(aQ, bK, vzero8());

            // 32-wide distance window covering clip(l-s,-128,128)+128 for this tile
            int w0 = lb - sb + 113;
            w0 = w0 < 0 ? 0 : (w0 > 225 ? 225 : w0);
            v16u aK  = fragA_e16(s_k, sb, lane);
            v16u be0 = fragB_e16(s_emb, w0, lane);
            v16u be1 = fragB_e16(s_emb, w0 + 16, lane);
            v8f qe0 = wmma_bf16(aQ, be0, vzero8());
            v8f qe1 = wmma_bf16(aQ, be1, vzero8());
            v8f ke0 = wmma_bf16(aK, be0, vzero8());
            v8f ke1 = wmma_bf16(aK, be1, vzero8());

#pragma unroll
            for (int r = 0; r < 8; ++r) {
                stg [(mrow + r) * 32 + col]      = qe0[r];
                stg [(mrow + r) * 32 + 16 + col] = qe1[r];
                stg2[(mrow + r) * 32 + col]      = ke0[r];
                stg2[(mrow + r) * 32 + 16 + col] = ke1[r];
            }
            asm volatile("s_wait_dscnt 0" ::: "memory");   // wave-local LDS RAW
#pragma unroll
            for (int r = 0; r < 8; ++r) {
                int l = lb + mrow + r;
                int s = sb + col;
                int d = l - s;
                d = d < -MM ? -MM : (d > MM ? MM : d);
                int idx = (d + MM) - w0;                    // 0..31 by construction
                S[st][r] = (acc[r] + stg[(mrow + r) * 32 + idx]
                                   + stg2[col * 32 + idx]) * scale;
            }
        }

        // ---- row softmax over 256 columns (rows live in 16-lane halves) ----
        float mx[8];
#pragma unroll
        for (int r = 0; r < 8; ++r) mx[r] = -3.4e38f;
#pragma unroll
        for (int st = 0; st < 16; ++st)
#pragma unroll
            for (int r = 0; r < 8; ++r) mx[r] = fmaxf(mx[r], S[st][r]);
#pragma unroll
        for (int dl = 1; dl < 16; dl <<= 1)
#pragma unroll
            for (int r = 0; r < 8; ++r) mx[r] = fmaxf(mx[r], __shfl_xor(mx[r], dl, 32));

        float sm[8];
#pragma unroll
        for (int r = 0; r < 8; ++r) sm[r] = 0.f;
#pragma unroll
        for (int st = 0; st < 16; ++st)
#pragma unroll
            for (int r = 0; r < 8; ++r) {
                float e = __expf(S[st][r] - mx[r]);
                S[st][r] = e;
                sm[r] += e;
            }
#pragma unroll
        for (int dl = 1; dl < 16; dl <<= 1)
#pragma unroll
            for (int r = 0; r < 8; ++r) sm[r] += __shfl_xor(sm[r], dl, 32);
#pragma unroll
        for (int r = 0; r < 8; ++r) sm[r] = 1.0f / sm[r];
#pragma unroll
        for (int st = 0; st < 16; ++st)
#pragma unroll
            for (int r = 0; r < 8; ++r) S[st][r] *= sm[r];

        // ---- O = P @ V, K=32 chunks via bf16 LDS re-staging ----
        v8f o = vzero8();
#pragma unroll
        for (int ch = 0; ch < 8; ++ch) {
#pragma unroll
            for (int r = 0; r < 8; ++r) {
                pstg[(mrow + r) * 32 + col]      = f2bf(S[2 * ch][r]);
                pstg[(mrow + r) * 32 + 16 + col] = f2bf(S[2 * ch + 1][r]);
            }
            asm volatile("s_wait_dscnt 0" ::: "memory");
            v16u aP = fragA_p(pstg, lane);
            v16u bV = fragB_v(s_v, ch * 32, lane);
            o = wmma_bf16(aP, bV, o);
        }

        // ---- write y in (f, n, c) layout ----
#pragma unroll
        for (int r = 0; r < 8; ++r) {
            int l = lb + mrow + r;
            y[((size_t)l * NN + n) * CC + h * EE + col] = o[r];
        }
    }
}

// ---------------------------------------------------------------------------
// Kernel 3: gated AddNorm + LN1. Block = 256 threads = 2 rows x 128 channels.
// ---------------------------------------------------------------------------
__global__ __launch_bounds__(256) void k_gate_ln(
    const float* __restrict__ y, const float* __restrict__ x,
    const float* __restrict__ Wb, const float* __restrict__ g1,
    const float* __restrict__ b1, float* __restrict__ h1f,
    unsigned short* __restrict__ h1b) {
    __shared__ float red[256];
    int tid = threadIdx.x;
    int rowi = tid >> 7, c = tid & 127;
    size_t r = (size_t)blockIdx.x * 2 + rowi;
    float yv = y[r * CC + c], xv = x[r * CC + c];

    // gate = sigmoid(Wb . [y, x, y-x])
    red[tid] = Wb[c] * yv + Wb[CC + c] * xv + Wb[2 * CC + c] * (yv - xv);
    __syncthreads();
    for (int s = 64; s > 0; s >>= 1) {
        if (c < s) red[tid] += red[tid + s];
        __syncthreads();
    }
    float gate = 1.0f / (1.0f + __expf(-red[rowi * 128]));
    __syncthreads();

    float z = gate * xv + (1.0f - gate) * yv;
    red[tid] = z;
    __syncthreads();
    for (int s = 64; s > 0; s >>= 1) {
        if (c < s) red[tid] += red[tid + s];
        __syncthreads();
    }
    float mean = red[rowi * 128] * (1.0f / CC);
    __syncthreads();
    float dz = z - mean;
    red[tid] = dz * dz;
    __syncthreads();
    for (int s = 64; s > 0; s >>= 1) {
        if (c < s) red[tid] += red[tid + s];
        __syncthreads();
    }
    float rstd = rsqrtf(red[rowi * 128] * (1.0f / CC) + 1e-5f);
    float h = dz * rstd * g1[c] + b1[c];
    h1f[r * CC + c] = h;
    h1b[r * CC + c] = f2bf(h);
}

// ---------------------------------------------------------------------------
// Kernel 4: FFN1 = GELU(h1 @ W1^T + b1), bf16 out. grid 6400, block 256.
// ---------------------------------------------------------------------------
__global__ __launch_bounds__(256) void k_ffn1(
    const unsigned short* __restrict__ h1b, const unsigned short* __restrict__ W1b,
    const float* __restrict__ b1, unsigned short* __restrict__ t) {
    int tid = threadIdx.x, lane = tid & 31, w = tid >> 5;
    int rb = blockIdx.x << 4;
    int ob = w << 4;
    v8f acc = vzero8();
#pragma unroll
    for (int kk = 0; kk < 4; ++kk) {
        v16u a = fragA_bf(h1b, CC, rb, kk * 32, lane);
        v16u b = fragB_bf(W1b, CC, ob, kk * 32, lane);
        acc = wmma_bf16(a, b, acc);
    }
    int mrow = (lane >> 4) << 3, col = lane & 15;
    int o = ob + col;
    float bias = b1[o];
#pragma unroll
    for (int r = 0; r < 8; ++r) {
        float v = acc[r] + bias;
        v = 0.5f * v * (1.0f + erff(v * 0.70710678118f));   // exact GELU
        t[(size_t)(rb + mrow + r) * CC + o] = f2bf(v);
    }
}

// ---------------------------------------------------------------------------
// Kernel 5: FFN2 + residual + LN2 -> out. Block = 256 (8 waves cover all 128
// cols of a 16-row stripe), LN fused through an LDS tile.
// ---------------------------------------------------------------------------
__global__ __launch_bounds__(256) void k_ffn2_ln(
    const unsigned short* __restrict__ t, const unsigned short* __restrict__ W2b,
    const float* __restrict__ b2, const float* __restrict__ h1f,
    const float* __restrict__ g2, const float* __restrict__ bln2,
    float* __restrict__ out) {
    __shared__ float tile[16 * 132];
    __shared__ float mu[16], rs[16];
    int tid = threadIdx.x, lane = tid & 31, w = tid >> 5;
    int rb = blockIdx.x << 4;
    int ob = w << 4;
    v8f acc = vzero8();
#pragma unroll
    for (int kk = 0; kk < 4; ++kk) {
        v16u a = fragA_bf(t, CC, rb, kk * 32, lane);
        v16u b = fragB_bf(W2b, CC, ob, kk * 32, lane);
        acc = wmma_bf16(a, b, acc);
    }
    int mrow = (lane >> 4) << 3, col = lane & 15;
    int o = ob + col;
#pragma unroll
    for (int r = 0; r < 8; ++r) {
        int m = mrow + r;
        tile[m * 132 + o] = acc[r] + b2[o] + h1f[(size_t)(rb + m) * CC + o];
    }
    __syncthreads();
    if (tid < 16) {
        float s = 0.f;
        for (int c = 0; c < CC; ++c) s += tile[tid * 132 + c];
        float m = s * (1.0f / CC);
        float v = 0.f;
        for (int c = 0; c < CC; ++c) {
            float d = tile[tid * 132 + c] - m;
            v += d * d;
        }
        mu[tid] = m;
        rs[tid] = rsqrtf(v * (1.0f / CC) + 1e-5f);
    }
    __syncthreads();
    for (int i = tid; i < 16 * CC; i += 256) {
        int m = i >> 7, c = i & 127;
        out[(size_t)(rb + m) * CC + c] =
            (tile[m * 132 + c] - mu[m]) * rs[m] * g2[c] + bln2[c];
    }
}

// ---------------------------------------------------------------------------
// Host launcher
// ---------------------------------------------------------------------------
extern "C" void kernel_launch(void* const* d_in, const int* in_sizes, int n_in,
                              void* d_out, int out_size, void* d_ws, size_t ws_size,
                              hipStream_t stream) {
    (void)in_sizes; (void)n_in; (void)out_size; (void)ws_size;
    const float* x    = (const float*)d_in[0];
    const float* tree = (const float*)d_in[1];
    const float* Wqkv = (const float*)d_in[2];
    const float* Wtk  = (const float*)d_in[3];
    const float* Wtv  = (const float*)d_in[4];
    const float* emb  = (const float*)d_in[5];
    const float* Wb   = (const float*)d_in[6];
    const float* ln1g = (const float*)d_in[7];
    const float* ln1b = (const float*)d_in[8];
    const float* ln2g = (const float*)d_in[9];
    const float* ln2b = (const float*)d_in[10];
    const float* W1   = (const float*)d_in[11];
    const float* b1   = (const float*)d_in[12];
    const float* W2   = (const float*)d_in[13];
    const float* b2   = (const float*)d_in[14];
    float* out = (float*)d_out;

    // workspace layout (256B aligned slabs)
    size_t off = 0;
    char* wsb = (char*)d_ws;
    auto alloc = [&](size_t bytes) -> void* {
        void* p = wsb + off;
        off += (bytes + 255) & ~(size_t)255;
        return p;
    };
    float* tk = (float*)alloc((size_t)NN * CC * 4);
    float* tv = (float*)alloc((size_t)NN * CC * 4);
    const size_t headElems = (size_t)NN * HH * FF * EE;       // 26.2 MB each (bf16)
    unsigned short* qg = (unsigned short*)alloc(headElems * 2);
    unsigned short* kg = (unsigned short*)alloc(headElems * 2);
    unsigned short* vg = (unsigned short*)alloc(headElems * 2);
    float* y   = (float*)alloc((size_t)ROWS * CC * 4);
    float* h1f = (float*)alloc((size_t)ROWS * CC * 4);
    unsigned short* h1b = (unsigned short*)alloc((size_t)ROWS * CC * 2);
    unsigned short* tb  = (unsigned short*)alloc((size_t)ROWS * CC * 2);
    unsigned short* xbb = (unsigned short*)alloc((size_t)ROWS * CC * 2);
    unsigned short* Wqb = (unsigned short*)alloc((size_t)3 * CC * CC * 2);
    unsigned short* W1b = (unsigned short*)alloc((size_t)CC * CC * 2);
    unsigned short* W2b = (unsigned short*)alloc((size_t)CC * CC * 2);
    unsigned short* emb_b = (unsigned short*)alloc((size_t)(2 * MM + 1) * EE * 2);

    // one-pass bf16 pre-conversion of all GEMM operands
    k_f2bf<<<(ROWS * CC + 255) / 256, 256, 0, stream>>>(x, xbb, ROWS * CC);
    k_f2bf<<<(3 * CC * CC + 255) / 256, 256, 0, stream>>>(Wqkv, Wqb, 3 * CC * CC);
    k_f2bf<<<(CC * CC + 255) / 256, 256, 0, stream>>>(W1, W1b, CC * CC);
    k_f2bf<<<(CC * CC + 255) / 256, 256, 0, stream>>>(W2, W2b, CC * CC);
    k_f2bf<<<((2 * MM + 1) * EE + 255) / 256, 256, 0, stream>>>(emb, emb_b, (2 * MM + 1) * EE);

    k_tree   <<<NN, 128, 0, stream>>>(tree, Wtk, Wtv, tk, tv);
    k_qkv    <<<ROWS / 16, 256, 0, stream>>>(xbb, Wqb, tk, tv, qg, kg, vg);
    k_attn   <<<NN * HH, 128, 0, stream>>>(qg, kg, vg, emb_b, y);
    k_gate_ln<<<ROWS / 2, 256, 0, stream>>>(y, x, Wb, ln1g, ln1b, h1f, h1b);
    k_ffn1   <<<ROWS / 16, 256, 0, stream>>>(h1b, W1b, b1, tb);
    k_ffn2_ln<<<ROWS / 16, 256, 0, stream>>>(tb, W2b, b2, h1f, ln2g, ln2b, out);
}
